// Metamorph_parameterReinforcer_90726889161002
// MI455X (gfx1250) — matrix operand order repaired and checked by reading, not compiled, not run on hardware
//
#include <hip/hip_runtime.h>
#include <hip/hip_bf16.h>
#include <math.h>
#include <stdint.h>

// ---------------------------------------------------------------------------
// B=512, MODES=1000, L=64. Spectral layers collapse analytically:
//   x_out[b,f,t] = Re(tanh( x_in[b,f,0] * T[t] )),
//   T[t] = sum_m (sum_n w[m,n]) * e^{+i 2 pi m t / 64}
// Dominant cost: lin1 GEMM (512x64000)x(64000x1000) in bf16 WMMA, with lin1_w
// converted fp32->bf16 on the fly (v_perm_b32 pack) while streaming 256 MB.
// A-tile staging uses the CDNA5 async global->LDS path (ASYNCcnt).
// ---------------------------------------------------------------------------

#define MODES 1000
#define LDIM  64
#define BATCH 512
#define KTOT  (MODES * LDIM)   // 64000

typedef __attribute__((ext_vector_type(16))) __bf16 v16bf;
typedef __attribute__((ext_vector_type(8)))  float  v8f;

// Pack two fp32 -> packed bf16 pair with round-to-nearest-even:
// one v_add3 per value + one v_perm_b32 for the pack.
static __device__ __forceinline__ unsigned pack2bf(float lo, float hi) {
  unsigned u0 = __float_as_uint(lo);
  unsigned u1 = __float_as_uint(hi);
  u0 += 0x7FFFu + ((u0 >> 16) & 1u);
  u1 += 0x7FFFu + ((u1 >> 16) & 1u);
  // bytes: [u1.b3, u1.b2, u0.b3, u0.b2]
  return __builtin_amdgcn_perm(u1, u0, 0x07060302u);
}

// Re(tanh(x + i*y)) = sinh(2x) / (cosh(2x) + cos(2y))
static __device__ __forceinline__ float retanh(float x, float y) {
  float x2 = fminf(fmaxf(2.0f * x, -60.0f), 60.0f);
  float y2 = 2.0f * y;
  float e  = __expf(x2);
  float ei = 1.0f / e;
  return (e - ei) / (e + ei + 2.0f * cosf(y2));
}

// ---------------------------------------------------------------------------
// Workspace float table F (after bf16 A matrix and out1):
//   F[  0.. 63] S2re   F[ 64..127] S2im      (row sums of wfft2)
//   F[128..191] p0re   F[192..255] p0im      (row sums of wfft0)
//   F[256..319] p1re   F[320..383] p1im      (row sums of wfft1)
//   F[384..447] T2re   F[448..511] T2im
//   F[512]=c0re F[513]=c0im F[514]=c1re F[515]=c1im
// ---------------------------------------------------------------------------

__global__ __launch_bounds__(256) void k_reduce(
    const float* __restrict__ w0re, const float* __restrict__ w0im,
    const float* __restrict__ w1re, const float* __restrict__ w1im,
    const float* __restrict__ w2re, const float* __restrict__ w2im,
    float* __restrict__ F) {
  __shared__ float sre[256], sim[256];
  const int bid = blockIdx.x;
  const int grp = bid >> 6;   // 0: wfft2, 1: wfft0, 2: wfft1
  const int row = bid & 63;
  const float* re = (grp == 0) ? w2re : (grp == 1) ? w0re : w1re;
  const float* im = (grp == 0) ? w2im : (grp == 1) ? w0im : w1im;
  float aR = 0.f, aI = 0.f;
  for (int i = threadIdx.x; i < MODES; i += 256) {
    aR += re[row * MODES + i];
    aI += im[row * MODES + i];
  }
  sre[threadIdx.x] = aR;
  sim[threadIdx.x] = aI;
  __syncthreads();
  for (int s = 128; s > 0; s >>= 1) {
    if (threadIdx.x < s) {
      sre[threadIdx.x] += sre[threadIdx.x + s];
      sim[threadIdx.x] += sim[threadIdx.x + s];
    }
    __syncthreads();
  }
  if (threadIdx.x == 0) {
    const int base = (grp == 0) ? 0 : (grp == 1) ? 128 : 256;
    F[base + row]      = sre[0];
    F[base + 64 + row] = sim[0];
  }
}

__global__ __launch_bounds__(64) void k_finalize(float* __restrict__ F) {
  const int t = threadIdx.x;
  float tr = 0.f, ti = 0.f;
  for (int m = 0; m < 64; ++m) {
    float ang = 6.283185307179586f * (float)((m * t) & 63) * (1.0f / 64.0f);
    float s, c;
    sincosf(ang, &s, &c);
    float Sr = F[m], Si = F[64 + m];
    tr += Sr * c - Si * s;
    ti += Sr * s + Si * c;
  }
  F[384 + t] = tr;
  F[448 + t] = ti;
  if (t == 0) {
    float c0r = 0.f, c0i = 0.f, c1r = 0.f, c1i = 0.f;
    for (int i = 0; i < 64; ++i) {
      c0r += F[128 + i]; c0i += F[192 + i];
      c1r += F[256 + i]; c1i += F[320 + i];
    }
    F[512] = c0r; F[513] = c0i; F[514] = c1r; F[515] = c1i;
  }
}

// Build X3 (GEMM A matrix) in bf16, row-major (512 x 64000).
__global__ __launch_bounds__(256) void k_build_a(
    const float* __restrict__ params, const float* __restrict__ F,
    unsigned short* __restrict__ A) {
  __shared__ float t2r[64], t2i[64];
  const int tid = threadIdx.x;
  if (tid < 64)        t2r[tid]      = F[384 + tid];
  else if (tid < 128)  t2i[tid - 64] = F[448 + (tid - 64)];
  __syncthreads();

  const int idx = blockIdx.x * 256 + tid;  // = b*1000 + f
  const float c0r = F[512], c0i = F[513], c1r = F[514], c1i = F[515];
  const float a0 = params[(size_t)idx * LDIM];
  const float a1 = retanh(a0 * c0r, a0 * c0i);
  const float a2 = retanh(a1 * c1r, a1 * c1i);

  const size_t base = (size_t)idx * LDIM;  // row b, k = f*64 + t
#pragma unroll
  for (int ch = 0; ch < 8; ++ch) {
    unsigned v[4];
#pragma unroll
    for (int j = 0; j < 4; ++j) {
      const int t0 = ch * 8 + j * 2;
      float f0 = retanh(a2 * t2r[t0],     a2 * t2i[t0]);
      float f1 = retanh(a2 * t2r[t0 + 1], a2 * t2i[t0 + 1]);
      v[j] = pack2bf(f0, f1);
    }
    *(uint4*)(A + base + ch * 8) = make_uint4(v[0], v[1], v[2], v[3]);
  }
}

// out1[b,n] = tanh( sum_k A[b,k]*W[n,k] + bias[n] )
// Workgroup: 256 threads = 8 waves, tile 32(M) x 128(N), K-step 64.
// Each wave: 2 accumulators (two 16-wide N subtiles) -> 4 WMMA per barrier
// pair, A fragment reused across the N subtiles. A tile staged with the
// CDNA5 async global->LDS copy (ASYNCcnt); W tile converted through VGPRs.
__global__ __launch_bounds__(256) void k_gemm1(
    const unsigned short* __restrict__ A, const float* __restrict__ W,
    const float* __restrict__ bias, float* __restrict__ out1) {
  __shared__ uint4 sA4[32][8];   // 32 batch rows x 64 bf16 (128B/row) = 4KB
  __shared__ uint4 sB4[128][8];  // 128 mode rows x 64 bf16 = 16KB

  const int tid  = threadIdx.x;
  const int lane = tid & 31;
  const int wave = tid >> 5;    // 0..7
  const int wm   = wave >> 2;   // 0..1
  const int wn   = wave & 3;    // 0..3
  const int mBase = blockIdx.x * 32;
  const int nBase = blockIdx.y * 128;

  // A staging: one 16B async copy per thread per iter.
  const int ar = tid >> 3, ac = tid & 7;
  const unsigned short* aSrc = A + (size_t)(mBase + ar) * KTOT + ac * 8;
  // Generic LDS pointers keep the byte offset in the low 32 bits
  // (aperture check truncates the address to those 32 bits).
  const unsigned aDstLds = (unsigned)(uintptr_t)&sA4[ar][ac];
  // B staging: thread -> (row bj, 32-float half bq). Clamp row: out-of-range
  // columns load valid-but-unused data; epilogue guard discards them.
  const int bj = tid >> 1, bq = tid & 1;
  const int gn  = nBase + bj;
  const int gnc = (gn < MODES) ? gn : (MODES - 1);
  const float* wSrc = W + (size_t)gnc * KTOT + bq * 32;

  // WMMA operand addressing (16-bit A 16x32 / B 32x16 lane striping):
  //   A lane: M = lane%16; lanes<16 take 16B chunks {0,1} of the 32-K window,
  //           lanes>=16 take {2,3}.
  //   B lane: N = lane%16; lanes<16 take chunks {0,1}, lanes>=16 {2,3}.
  const int rowA  = wm * 16 + (lane & 15);
  const int rowB0 = wn * 32 + (lane & 15);
  const int rowB1 = rowB0 + 16;
  const int hi    = lane >> 4;

  v8f acc0 = {0.f, 0.f, 0.f, 0.f, 0.f, 0.f, 0.f, 0.f};
  v8f acc1 = {0.f, 0.f, 0.f, 0.f, 0.f, 0.f, 0.f, 0.f};

  for (int k0 = 0; k0 < KTOT; k0 += 64) {
    // stage A tile: 32x64 bf16 = 4KB, async global->LDS (no VGPR round trip)
    {
      unsigned long long gaddr = (unsigned long long)(uintptr_t)(aSrc + k0);
      asm volatile("global_load_async_to_lds_b128 %0, %1, off"
                   :: "v"(aDstLds), "v"(gaddr) : "memory");
    }
    // stage W tile: 128x64 fp32 -> bf16 (8 float4 loads + perm packs)
    {
      const float4* p = (const float4*)(wSrc + k0);
#pragma unroll
      for (int j = 0; j < 4; ++j) {
        float4 f0 = p[2 * j];
        float4 f1 = p[2 * j + 1];
        sB4[bj][bq * 4 + j] =
            make_uint4(pack2bf(f0.x, f0.y), pack2bf(f0.z, f0.w),
                       pack2bf(f1.x, f1.y), pack2bf(f1.z, f1.w));
      }
    }
    asm volatile("s_wait_asynccnt 0" ::: "memory");
    __syncthreads();

#pragma unroll
    for (int ks = 0; ks < 2; ++ks) {
      const int cb = ks * 4;  // uint4 index of this 32-K window
      union { uint4 u[2]; v16bf v; } av, bv0, bv1;
      av.u[0]  = sA4[rowA][cb + 2 * hi];
      av.u[1]  = sA4[rowA][cb + 2 * hi + 1];
      bv0.u[0] = sB4[rowB0][cb + 2 * hi];
      bv0.u[1] = sB4[rowB0][cb + 2 * hi + 1];
      bv1.u[0] = sB4[rowB1][cb + 2 * hi];
      bv1.u[1] = sB4[rowB1][cb + 2 * hi + 1];
      acc0 = __builtin_amdgcn_wmma_f32_16x16x32_bf16(
          false, av.v, false, bv0.v, (short)0, acc0, false, false);
      acc1 = __builtin_amdgcn_wmma_f32_16x16x32_bf16(
          false, av.v, false, bv1.v, (short)0, acc1, false, false);
    }
    __syncthreads();
  }

  // Epilogue: D layout: VGPR r holds M = r + 8*(lane>=16), N = lane%16.
  const int rbase = mBase + wm * 16 + hi * 8;
#pragma unroll
  for (int nt = 0; nt < 2; ++nt) {
    const int n = nBase + wn * 32 + nt * 16 + (lane & 15);
    if (n < MODES) {
      const float bn = bias[n];
      const v8f& acc = nt ? acc1 : acc0;
#pragma unroll
      for (int r = 0; r < 8; ++r) {
        out1[(size_t)(rbase + r) * MODES + n] = tanhf(acc[r] + bn);
      }
    }
  }
}

// out[b,l] = sigmoid( sum_m out1[b,m]*W2[l,m] + b2[l] ). 65 MFLOP, tiny.
__global__ __launch_bounds__(64) void k_gemm2(
    const float* __restrict__ out1, const float* __restrict__ W2,
    const float* __restrict__ b2, float* __restrict__ out) {
  const int b = blockIdx.x, l = threadIdx.x;
  const float* xr = out1 + (size_t)b * MODES;
  const float* wr = W2 + (size_t)l * MODES;
  float acc = 0.f;
  for (int m = 0; m < MODES; m += 4) {
    acc = fmaf(xr[m],     wr[m],     acc);
    acc = fmaf(xr[m + 1], wr[m + 1], acc);
    acc = fmaf(xr[m + 2], wr[m + 2], acc);
    acc = fmaf(xr[m + 3], wr[m + 3], acc);
  }
  acc += b2[l];
  out[(size_t)b * LDIM + l] = 1.0f / (1.0f + __expf(-acc));
}

extern "C" void kernel_launch(void* const* d_in, const int* in_sizes, int n_in,
                              void* d_out, int out_size, void* d_ws,
                              size_t ws_size, hipStream_t stream) {
  (void)in_sizes; (void)n_in; (void)out_size; (void)ws_size;
  const float* params = (const float*)d_in[0];
  const float* w0re   = (const float*)d_in[1];
  const float* w0im   = (const float*)d_in[2];
  const float* w1re   = (const float*)d_in[3];
  const float* w1im   = (const float*)d_in[4];
  const float* w2re   = (const float*)d_in[5];
  const float* w2im   = (const float*)d_in[6];
  const float* lin1w  = (const float*)d_in[7];
  const float* lin1b  = (const float*)d_in[8];
  const float* lin2w  = (const float*)d_in[9];
  const float* lin2b  = (const float*)d_in[10];
  float* out = (float*)d_out;

  const size_t A_BYTES    = (size_t)BATCH * KTOT * 2;   // 65,536,000 (bf16 X3)
  const size_t OUT1_BYTES = (size_t)BATCH * MODES * 4;  //  2,048,000
  unsigned short* Abf = (unsigned short*)d_ws;
  float* out1 = (float*)((char*)d_ws + A_BYTES);
  float* F    = (float*)((char*)d_ws + A_BYTES + OUT1_BYTES);

  k_reduce<<<192, 256, 0, stream>>>(w0re, w0im, w1re, w1im, w2re, w2im, F);
  k_finalize<<<1, 64, 0, stream>>>(F);
  k_build_a<<<(BATCH * MODES) / 256, 256, 0, stream>>>(params, F, Abf);
  dim3 g1(BATCH / 32, 8);  // 32x128 tiles; N padded to 1024, guarded
  k_gemm1<<<g1, 256, 0, stream>>>(Abf, lin1w, lin1b, out1);
  k_gemm2<<<BATCH, 64, 0, stream>>>(out1, lin2w, lin2b, out);
}
